// ActionHeadLayer_79585743995158
// MI455X (gfx1250) — compile-verified, matrix operand order checked by reading, not run on hardware
//
#include <hip/hip_runtime.h>
#include <hip/hip_bf16.h>
#include <math.h>

// ---------------- problem constants ----------------
#define BB      8
#define NN      2048
#define MM      73          // 1 + 8 + 64
#define DI      512
#define HEADS   8
#define DH      64
#define JTOT    (NN + MM)   // 2121
#define ROWS    (BB * MM)   // 584
#define SKV_ROWS (BB * 2 * NN) // 32768

typedef __attribute__((ext_vector_type(16))) __bf16 v16bf;
typedef __attribute__((ext_vector_type(8)))  __bf16 v8bf;
typedef __attribute__((ext_vector_type(8)))  float  v8f_t;

// ---------------- CDNA5 async global->LDS (ASYNCcnt path) ----------------
#if defined(__has_builtin)
#if __has_builtin(__builtin_amdgcn_global_load_async_to_lds_b128)
#define HAVE_ASYNC_LDS 1
#endif
#endif

#ifdef HAVE_ASYNC_LDS
typedef __attribute__((ext_vector_type(4))) int v4i_t;
typedef __attribute__((address_space(1))) v4i_t g_v4i;
typedef __attribute__((address_space(3))) v4i_t l_v4i;
__device__ __forceinline__ void async_copy16(const float* g, float* l) {
    __builtin_amdgcn_global_load_async_to_lds_b128(
        (g_v4i*)(unsigned long long)g,
        (l_v4i*)(unsigned int)(unsigned long long)l,
        0, 0);
}
__device__ __forceinline__ void async_wait0() {
#if __has_builtin(__builtin_amdgcn_s_wait_asynccnt)
    __builtin_amdgcn_s_wait_asynccnt(0);
#else
    asm volatile("s_wait_asynccnt 0" ::: "memory");
#endif
}
#endif

// =====================================================================
// Kernel 1: big GEMM  skv = [vlm_keys ; vlm_values] @ w_skv + b_skv
//   A: (B*2N, 2048) fp32 (two source tensors), W: (2048, 512) fp32
//   bf16 WMMA 16x16x32, f32 accumulate.
//   Workgroup: 256 threads = 8 waves (4M x 2N), tile 64(M) x 64(N),
//   K-step 32, each wave does 2 WMMAs per step (A fragment reused).
//   A tile staged fp32 in LDS via GLOBAL_LOAD_ASYNC_TO_LDS_B128,
//   converted to bf16 at fragment build; B tile converted on store.
// =====================================================================
__global__ __launch_bounds__(256)
void skv_gemm_wmma(const float* __restrict__ vk,
                   const float* __restrict__ vv,
                   const float* __restrict__ w,     // (2048, 512)
                   const float* __restrict__ bias,  // (512)
                   float* __restrict__ out)         // (32768, 512)
{
    __shared__ __align__(16) float  Asf[64][32];  // fp32 (m, k)  8KB
    __shared__ __align__(16) __bf16 Bs[64][32];   // TRANSPOSED bf16 (n, k) 4KB

    const int m0   = blockIdx.x * 64;
    const int n0   = blockIdx.y * 64;
    const int tid  = threadIdx.x;
    const int wave = tid >> 5;
    const int lane = tid & 31;
    const int wm   = (wave >> 1) * 16;   // 0,16,32,48
    const int wn   = (wave & 1) * 32;    // 0,32
    const int half = lane >> 4;          // 0: K 0-7/16-23, 1: K 8-15/24-31
    const int lr   = lane & 15;

    // A-load mapping: 64x32 f32 = 2048 elems / 256 thr = 8 each
    const int arow = tid >> 2;           // 0..63
    const int akk  = (tid & 3) * 8;      // 0,8,16,24
    const size_t rg = (size_t)m0 + arow;
    const int    bbx = (int)(rg >> 12);  // / 4096
    const int    rr  = (int)(rg & 4095);
    const float* arow_ptr = (rr < NN)
        ? vk + (((size_t)bbx << 11) + rr) * 2048
        : vv + (((size_t)bbx << 11) + (rr - NN)) * 2048;
    float* alds = &Asf[arow][akk];

    // B-load mapping: 64(n) x 32(k) = 2048 elems / 256 thr = 8 each
    const int bn  = tid >> 2;            // 0..63 (n within tile)
    const int bk  = (tid & 3) * 8;       // 0,8,16,24

    v8f_t acc0 = {};
    v8f_t acc1 = {};
    const float* afp  = &Asf[wm + lr][0];
    const v8bf* bptr0 = (const v8bf*)(&Bs[wn + lr][0]);
    const v8bf* bptr1 = (const v8bf*)(&Bs[wn + 16 + lr][0]);

    for (int kt = 0; kt < 64; ++kt) {
        const int k0 = kt * 32;
        const float* aga = arow_ptr + k0 + akk;
        // ---- stage B fp32 -> registers (strided columns of w) ----
        float wb[8];
        #pragma unroll
        for (int i = 0; i < 8; ++i)
            wb[i] = w[(size_t)(k0 + bk + i) * DI + n0 + bn];
        if (kt + 1 < 64) {
            __builtin_prefetch(aga + 32, 0, 1);
        }
        __syncthreads();               // previous iteration's LDS reads done
        // ---- A: async DMA global -> LDS (fp32) ----
#ifdef HAVE_ASYNC_LDS
        async_copy16(aga,     alds);
        async_copy16(aga + 4, alds + 4);
#else
        *(float4*)(alds)     = *(const float4*)(aga);
        *(float4*)(alds + 4) = *(const float4*)(aga + 4);
#endif
        // ---- B: convert to bf16, store transposed ----
        #pragma unroll
        for (int i = 0; i < 8; ++i)
            Bs[bn][bk + i] = (__bf16)wb[i];
#ifdef HAVE_ASYNC_LDS
        async_wait0();
#endif
        __syncthreads();
        // ---- A fragment: fp32 LDS -> bf16 regs (ISA 16-bit A layout) ----
        float4 a0 = *(const float4*)(afp + half * 8);
        float4 a1 = *(const float4*)(afp + half * 8 + 4);
        float4 a2 = *(const float4*)(afp + 16 + half * 8);
        float4 a3 = *(const float4*)(afp + 16 + half * 8 + 4);
        v16bf afr;
        afr[0]  = (__bf16)a0.x; afr[1]  = (__bf16)a0.y;
        afr[2]  = (__bf16)a0.z; afr[3]  = (__bf16)a0.w;
        afr[4]  = (__bf16)a1.x; afr[5]  = (__bf16)a1.y;
        afr[6]  = (__bf16)a1.z; afr[7]  = (__bf16)a1.w;
        afr[8]  = (__bf16)a2.x; afr[9]  = (__bf16)a2.y;
        afr[10] = (__bf16)a2.z; afr[11] = (__bf16)a2.w;
        afr[12] = (__bf16)a3.x; afr[13] = (__bf16)a3.y;
        afr[14] = (__bf16)a3.z; afr[15] = (__bf16)a3.w;
        // ---- B fragments (bf16 directly from LDS) ----
        v8bf b0lo = bptr0[half];
        v8bf b0hi = bptr0[2 + half];
        v8bf b1lo = bptr1[half];
        v8bf b1hi = bptr1[2 + half];
        v16bf bfr0 = __builtin_shufflevector(b0lo, b0hi,
            0,1,2,3,4,5,6,7,8,9,10,11,12,13,14,15);
        v16bf bfr1 = __builtin_shufflevector(b1lo, b1hi,
            0,1,2,3,4,5,6,7,8,9,10,11,12,13,14,15);
        acc0 = __builtin_amdgcn_wmma_f32_16x16x32_bf16(
            false, afr, false, bfr0, (short)0, acc0, false, false);
        acc1 = __builtin_amdgcn_wmma_f32_16x16x32_bf16(
            false, afr, false, bfr1, (short)0, acc1, false, false);
    }

    // ---- epilogue: C layout = 8 VGPRs, lanes 0-15 -> M 0-7, 16-31 -> M 8-15
    const int m_base = m0 + wm + half * 8;
    const int col0   = n0 + wn + lr;
    const int col1   = col0 + 16;
    const float bv0 = bias[col0];
    const float bv1 = bias[col1];
    #pragma unroll
    for (int i = 0; i < 8; ++i) {
        out[(size_t)(m_base + i) * DI + col0] = acc0[i] + bv0;
        out[(size_t)(m_base + i) * DI + col1] = acc1[i] + bv1;
    }
}

// =====================================================================
// Kernel 2: gather orig = concat(t_tokens, proprio, action)  (B, 73, 512)
// =====================================================================
__global__ void gather_orig_k(const float* __restrict__ t,
                              const float* __restrict__ p,
                              const float* __restrict__ a,
                              float* __restrict__ orig)
{
    int idx = blockIdx.x * 256 + threadIdx.x;
    if (idx >= ROWS * DI) return;
    int d   = idx & (DI - 1);
    int rt  = idx >> 9;
    int b   = rt / MM;
    int tok = rt % MM;
    float v;
    if (tok == 0)      v = t[(size_t)b * DI + d];
    else if (tok < 9)  v = p[((size_t)b * 8 + (tok - 1)) * DI + d];
    else               v = a[((size_t)b * 64 + (tok - 9)) * DI + d];
    orig[idx] = v;
}

// =====================================================================
// Kernel 3: rmsnorm rows of (R, 512)
// =====================================================================
__global__ __launch_bounds__(256)
void rmsnorm_k(const float* __restrict__ X,
               const float* __restrict__ w,
               float* __restrict__ Y)
{
    __shared__ float red[256];
    const int r = blockIdx.x;
    const float* x = X + (size_t)r * DI;
    float ss = 0.f;
    for (int i = threadIdx.x; i < DI; i += 256) { float v = x[i]; ss = fmaf(v, v, ss); }
    red[threadIdx.x] = ss;
    __syncthreads();
    for (int s = 128; s > 0; s >>= 1) {
        if (threadIdx.x < s) red[threadIdx.x] += red[threadIdx.x + s];
        __syncthreads();
    }
    const float scale = rsqrtf(red[0] * (1.0f / DI) + 1e-6f);
    for (int i = threadIdx.x; i < DI; i += 256)
        Y[(size_t)r * DI + i] = x[i] * scale * (1.0f + w[i]);
}

// =====================================================================
// Kernel 4: small GEMM  C[r,n] = act(A[r,:] @ W[:,n] + bias[n])
//   mode 0: identity; 1: exact GELU; 2: gate (C = Arow[n] * sigmoid(v))
// =====================================================================
__global__ __launch_bounds__(256)
void small_gemm_k(const float* __restrict__ A,
                  const float* __restrict__ W,
                  const float* __restrict__ bias,
                  float* __restrict__ C,
                  int K, int Nout, int mode)
{
    __shared__ float row[DI];
    const int r = blockIdx.x;
    const float* a = A + (size_t)r * K;
    for (int i = threadIdx.x; i < K; i += 256) row[i] = a[i];
    __syncthreads();
    for (int n = threadIdx.x; n < Nout; n += 256) {
        const float* wcol = W + n;
        float acc = bias[n];
        #pragma unroll 4
        for (int k = 0; k < K; ++k)
            acc = fmaf(row[k], wcol[(size_t)k * Nout], acc);
        float o;
        if (mode == 1) {
            o = 0.5f * acc * (1.0f + erff(acc * 0.70710678118654752f));
        } else if (mode == 2) {
            o = row[n] * (1.0f / (1.0f + __expf(-acc)));
        } else {
            o = acc;
        }
        C[(size_t)r * Nout + n] = o;
    }
}

// =====================================================================
// Kernel 5: RoPE on q  -> Qr (B, H, 73, 64), pos = 2048 + i
// =====================================================================
__global__ void rope_q_k(const float* __restrict__ qkv,  // (584, 1536)
                         float* __restrict__ Qr)
{
    int idx = blockIdx.x * 256 + threadIdx.x;
    if (idx >= BB * HEADS * MM * DH) return;
    int d    = idx & (DH - 1);
    int rest = idx >> 6;
    int i    = rest % MM;
    int h    = (rest / MM) & (HEADS - 1);
    int b    = rest / (MM * HEADS);
    const float* src = qkv + ((size_t)(b * MM + i)) * 1536 + h * DH;
    float x = src[d];
    float y = src[d ^ 32];
    int   ii  = d & 31;
    float inv = __powf(10000.0f, -(float)ii * (1.0f / 32.0f));
    float ang = (float)(NN + i) * inv;
    float c = __cosf(ang), s = __sinf(ang);
    float o = (d < 32) ? (x * c - y * s) : (x * c + y * s);
    Qr[((size_t)(b * HEADS + h) * MM + i) * DH + d] = o;
}

// =====================================================================
// Kernel 6: RoPE on keys -> Kr (B, H, 2121, 64), pos = j
//   j < 2048: sk from skv rows [b*4096 + j]; else ak from qkv cols 512..1023
// =====================================================================
__global__ void rope_k_k(const float* __restrict__ skv,  // (32768, 512)
                         const float* __restrict__ qkv,  // (584, 1536)
                         float* __restrict__ Kr)
{
    int idx = blockIdx.x * 256 + threadIdx.x;
    if (idx >= BB * HEADS * JTOT * DH) return;
    int d    = idx & (DH - 1);
    int rest = idx >> 6;
    int j    = rest % JTOT;
    int h    = (rest / JTOT) & (HEADS - 1);
    int b    = rest / (JTOT * HEADS);
    const float* src = (j < NN)
        ? skv + ((size_t)b * 4096 + j) * DI + h * DH
        : qkv + ((size_t)(b * MM + (j - NN))) * 1536 + DI + h * DH;
    float x = src[d];
    float y = src[d ^ 32];
    int   ii  = d & 31;
    float inv = __powf(10000.0f, -(float)ii * (1.0f / 32.0f));
    float ang = (float)j * inv;
    float c = __cosf(ang), s = __sinf(ang);
    float o = (d < 32) ? (x * c - y * s) : (x * c + y * s);
    Kr[((size_t)(b * HEADS + h) * JTOT + j) * DH + d] = o;
}

// =====================================================================
// Kernel 7: attention, one block per (b, h, i), two-pass softmax.
//   Mask is a contiguous prefix: JL = 2048 + (i==0 ? 1 : (i<9 ? 9 : 73))
// =====================================================================
__global__ __launch_bounds__(128)
void attn_k(const float* __restrict__ Qr,
            const float* __restrict__ Kr,
            const float* __restrict__ skv,   // values (sv) rows b*4096+2048+j
            const float* __restrict__ qkv,   // values (av) cols 1024..1535
            float* __restrict__ attn)        // (584, 512)
{
    __shared__ float sc[JTOT + 7];
    __shared__ float qrow[DH];
    __shared__ float red[128];
    const int bid = blockIdx.x;
    const int i   = bid % MM;
    const int h   = (bid / MM) & (HEADS - 1);
    const int b   = bid / (MM * HEADS);
    const int JL  = NN + (i == 0 ? 1 : (i < 9 ? 9 : MM));

    const float* q = Qr + ((size_t)(b * HEADS + h) * MM + i) * DH;
    if (threadIdx.x < DH) qrow[threadIdx.x] = q[threadIdx.x];
    __syncthreads();

    const float* kb = Kr + (size_t)(b * HEADS + h) * JTOT * DH;
    float lmax = -1e30f;
    for (int j = threadIdx.x; j < JL; j += 128) {
        const float* kj = kb + (size_t)j * DH;
        float s = 0.f;
        #pragma unroll
        for (int d = 0; d < DH; ++d) s = fmaf(qrow[d], kj[d], s);
        s *= 0.125f;                  // DH^-0.5
        sc[j] = s;
        lmax = fmaxf(lmax, s);
    }
    red[threadIdx.x] = lmax;
    __syncthreads();
    for (int st = 64; st > 0; st >>= 1) {
        if (threadIdx.x < st)
            red[threadIdx.x] = fmaxf(red[threadIdx.x], red[threadIdx.x + st]);
        __syncthreads();
    }
    const float mx = red[0];
    __syncthreads();

    float lsum = 0.f;
    for (int j = threadIdx.x; j < JL; j += 128) {
        float p = __expf(sc[j] - mx);
        sc[j] = p;
        lsum += p;
    }
    red[threadIdx.x] = lsum;
    __syncthreads();
    for (int st = 64; st > 0; st >>= 1) {
        if (threadIdx.x < st) red[threadIdx.x] += red[threadIdx.x + st];
        __syncthreads();
    }
    const float invsum = 1.0f / red[0];

    if (threadIdx.x < DH) {
        const int d = threadIdx.x;
        float acc = 0.f;
        for (int j = 0; j < JL; ++j) {
            float v = (j < NN)
                ? skv[((size_t)b * 4096 + NN + j) * DI + h * DH + d]
                : qkv[((size_t)(b * MM + (j - NN))) * 1536 + 2 * DI + h * DH + d];
            acc = fmaf(sc[j], v, acc);
        }
        attn[((size_t)(b * MM + i)) * DI + h * DH + d] = acc * invsum;
    }
}

// =====================================================================
// Kernel 8: elementwise add  c = a + b
// =====================================================================
__global__ void add_k(const float* __restrict__ a,
                      const float* __restrict__ b,
                      float* __restrict__ c, int n)
{
    int idx = blockIdx.x * 256 + threadIdx.x;
    if (idx < n) c[idx] = a[idx] + b[idx];
}

// =====================================================================
// Kernel 9: final residual + tuple scatter into d_out
//   out = concat_flat( h[:,0:1], h[:,1:9], h[:,9:73] )
// =====================================================================
__global__ void final_scatter_k(const float* __restrict__ h,
                                const float* __restrict__ f2,
                                float* __restrict__ out)
{
    int idx = blockIdx.x * 256 + threadIdx.x;
    if (idx >= ROWS * DI) return;
    float v = h[idx] + f2[idx];
    int d   = idx & (DI - 1);
    int rt  = idx >> 9;
    int b   = rt / MM;
    int tok = rt % MM;
    size_t o;
    if (tok == 0)
        o = (size_t)b * DI + d;
    else if (tok < 9)
        o = (size_t)BB * DI + ((size_t)(b * 8 + (tok - 1))) * DI + d;
    else
        o = (size_t)BB * DI + (size_t)BB * 8 * DI
          + ((size_t)(b * 64 + (tok - 9))) * DI + d;
    out[o] = v;
}

// =====================================================================
// launch
// =====================================================================
extern "C" void kernel_launch(void* const* d_in, const int* in_sizes, int n_in,
                              void* d_out, int out_size, void* d_ws, size_t ws_size,
                              hipStream_t stream)
{
    (void)in_sizes; (void)n_in; (void)out_size; (void)ws_size;
    const float* vk      = (const float*)d_in[0];
    const float* vv      = (const float*)d_in[1];
    const float* t_tok   = (const float*)d_in[2];
    const float* p_tok   = (const float*)d_in[3];
    const float* a_tok   = (const float*)d_in[4];
    const float* w_skv   = (const float*)d_in[5];
    const float* b_skv   = (const float*)d_in[6];
    const float* w_qkv   = (const float*)d_in[7];
    const float* b_qkv   = (const float*)d_in[8];
    const float* w_gate  = (const float*)d_in[9];
    const float* b_gate  = (const float*)d_in[10];
    const float* w_out   = (const float*)d_in[11];
    const float* b_out   = (const float*)d_in[12];
    const float* w_ffn1  = (const float*)d_in[13];
    const float* b_ffn1  = (const float*)d_in[14];
    const float* w_ffn2  = (const float*)d_in[15];
    const float* b_ffn2  = (const float*)d_in[16];
    const float* norm1_w = (const float*)d_in[17];
    const float* norm2_w = (const float*)d_in[18];
    float* out = (float*)d_out;

    // workspace layout (floats)
    float* W    = (float*)d_ws;
    float* skv  = W;                                  // 32768*512 = 16,777,216
    float* Kr   = skv + (size_t)SKV_ROWS * DI;        //  8,687,616
    float* orig = Kr  + (size_t)BB * HEADS * JTOT * DH;
    float* xn   = orig + (size_t)ROWS * DI;
    float* qkv  = xn   + (size_t)ROWS * DI;           // 584*1536
    float* Qr   = qkv  + (size_t)ROWS * 1536;
    float* attn = Qr   + (size_t)BB * HEADS * MM * DH;
    float* gatd = attn + (size_t)ROWS * DI;
    float* proj = gatd + (size_t)ROWS * DI;
    float* hb   = proj + (size_t)ROWS * DI;
    float* x2   = hb   + (size_t)ROWS * DI;
    float* f1   = x2   + (size_t)ROWS * DI;
    float* f2   = f1   + (size_t)ROWS * DI;

    const int elems = ROWS * DI;                       // 299,008
    const int eb    = (elems + 255) / 256;

    // 1) big WMMA GEMM: skv
    dim3 g1(SKV_ROWS / 64, DI / 64);
    skv_gemm_wmma<<<g1, 256, 0, stream>>>(vk, vv, w_skv, b_skv, skv);

    // 2) token path
    gather_orig_k<<<eb, 256, 0, stream>>>(t_tok, p_tok, a_tok, orig);
    rmsnorm_k<<<ROWS, 256, 0, stream>>>(orig, norm1_w, xn);
    small_gemm_k<<<ROWS, 256, 0, stream>>>(xn, w_qkv, b_qkv, qkv, DI, 3 * DI, 0);

    // 3) RoPE
    rope_q_k<<<(BB * HEADS * MM * DH + 255) / 256, 256, 0, stream>>>(qkv, Qr);
    rope_k_k<<<(BB * HEADS * JTOT * DH + 255) / 256, 256, 0, stream>>>(skv, qkv, Kr);

    // 4) attention
    attn_k<<<BB * HEADS * MM, 128, 0, stream>>>(Qr, Kr, skv, qkv, attn);

    // 5) gate + out proj + residual
    small_gemm_k<<<ROWS, 256, 0, stream>>>(attn, w_gate, b_gate, gatd, DI, DI, 2);
    small_gemm_k<<<ROWS, 256, 0, stream>>>(gatd, w_out, b_out, proj, DI, DI, 0);
    add_k<<<eb, 256, 0, stream>>>(orig, proj, hb, elems);

    // 6) FFN + final residual/scatter
    rmsnorm_k<<<ROWS, 256, 0, stream>>>(hb, norm2_w, x2);
    small_gemm_k<<<ROWS, 256, 0, stream>>>(x2, w_ffn1, b_ffn1, f1, DI, DI, 1);
    small_gemm_k<<<ROWS, 256, 0, stream>>>(f1, w_ffn2, b_ffn2, f2, DI, DI, 0);
    final_scatter_k<<<eb, 256, 0, stream>>>(hb, f2, out);
}
